// SASRecEncoder_49460843381031
// MI455X (gfx1250) — compile-verified, matrix operand order checked by reading, not run on hardware
//
#include <hip/hip_runtime.h>
#include <hip/hip_bf16.h>
#include <math.h>

typedef __attribute__((ext_vector_type(16))) _Float16 v16h;
typedef __attribute__((ext_vector_type(8)))  _Float16 v8h;
typedef __attribute__((ext_vector_type(8)))  float    v8f;
typedef __attribute__((ext_vector_type(4)))  unsigned int u32x4;
typedef __attribute__((ext_vector_type(8)))  unsigned int u32x8;

#define B_   512
#define L_   200
#define H_   512
#define NL_  4
#define NH_  8
#define DH_  64
#define H3_  1536
#define H4_  2048
#define BL_  (B_ * L_)
#define QTILES 13   // ceil(200/16)

// ---------------------------------------------------------------------------
// WMMA / TDM helpers
// ---------------------------------------------------------------------------
__device__ __forceinline__ v8f wmma32(v16h a, v16h b, v8f c) {
  return __builtin_amdgcn_wmma_f32_16x16x32_f16(
      /*neg_a=*/false, a, /*neg_b=*/false, b,
      /*c_mod=*/(short)0, c, /*reuse_a=*/false, /*reuse_b=*/false);
}

// Load one 16x32 f16 A/B fragment. `base` points at (row_start + k0) of a
// row-major matrix (row fixed per lane already folded in by caller).
// ISA 16-bit A layout: lanes 0-15 -> K {0..7,16..23}; lanes 16-31 -> K {8..15,24..31}.
__device__ __forceinline__ v16h frag_ld(const _Float16* base) {
  const int lane = threadIdx.x & 31;
  const int kb = (lane & 16) ? 8 : 0;
  v8h lo = *(const v8h*)(base + kb);
  v8h hi = *(const v8h*)(base + kb + 16);
  v16h r;
#pragma unroll
  for (int i = 0; i < 8; ++i) { r[i] = lo[i]; r[i + 8] = hi[i]; }
  return r;
}

// Tensor Data Mover: async DMA of a 2D tile (global -> LDS). D# built per
// CDNA5 ISA; 2-group form (VADDR2/3 = NULL, 2D tensor).
__device__ __forceinline__ void tdm_load_to_lds(u32x4 g0, u32x8 g1) {
  asm volatile("tensor_load_to_lds %0, %1" : : "s"(g0), "s"(g1) : "memory");
}

// Build group0: count=1, lds byte offset, 57-bit global byte address, type=2.
__device__ __forceinline__ u32x4 tdm_g0(unsigned lds_off, const void* gptr) {
  unsigned long long ga = (unsigned long long)(size_t)gptr;
  u32x4 g0;
  g0[0] = 1u;                                   // count = 1 (valid descriptor)
  g0[1] = lds_off;                              // lds_addr (bytes)
  g0[2] = (unsigned)ga;                         // global_addr[31:0]
  g0[3] = (unsigned)(ga >> 32) | (2u << 30);    // global_addr[56:32] | type=2
  return g0;
}

// ---------------------------------------------------------------------------
// Embedding: x[b,l,:] = item_embed[hist[b,l]] + pos_embed[l]
// ---------------------------------------------------------------------------
__global__ __launch_bounds__(128) void embed_kernel(
    const int* __restrict__ hist, const float* __restrict__ emb,
    const float* __restrict__ pos, float* __restrict__ x) {
  const int row = blockIdx.x;            // b*L + l
  const int l = row % L_;
  const int id = hist[row];
  const float4* e = (const float4*)(emb + (size_t)id * H_);
  const float4* p = (const float4*)(pos + (size_t)l * H_);
  float4* o = (float4*)(x + (size_t)row * H_);
  for (int c = threadIdx.x; c < H_ / 4; c += 128) {
    float4 ev = e[c], pv = p[c];
    float4 r; r.x = ev.x + pv.x; r.y = ev.y + pv.y;
    r.z = ev.z + pv.z; r.w = ev.w + pv.w;
    o[c] = r;
  }
}

// ---------------------------------------------------------------------------
// LayerNorm: one wave per row (H=512 -> 16 floats/lane), f32 in -> f16 out
// ---------------------------------------------------------------------------
__global__ __launch_bounds__(256) void ln_kernel(
    const float* __restrict__ x, const float* __restrict__ g,
    const float* __restrict__ bt, _Float16* __restrict__ out, int rows) {
  const int lane = threadIdx.x & 31;
  const int row = blockIdx.x * 8 + (threadIdx.x >> 5);
  if (row >= rows) return;
  const float4* xp = (const float4*)(x + (size_t)row * H_);
  float4 v[4];
  float s = 0.f;
#pragma unroll
  for (int i = 0; i < 4; ++i) {
    v[i] = xp[lane + 32 * i];
    s += v[i].x + v[i].y + v[i].z + v[i].w;
  }
#pragma unroll
  for (int m = 16; m >= 1; m >>= 1) s += __shfl_xor(s, m, 32);
  const float mean = s * (1.0f / H_);
  float vs = 0.f;
#pragma unroll
  for (int i = 0; i < 4; ++i) {
    float dx = v[i].x - mean, dy = v[i].y - mean;
    float dz = v[i].z - mean, dw = v[i].w - mean;
    vs += dx * dx + dy * dy + dz * dz + dw * dw;
  }
#pragma unroll
  for (int m = 16; m >= 1; m >>= 1) vs += __shfl_xor(vs, m, 32);
  const float inv = rsqrtf(vs * (1.0f / H_) + 1e-5f);
  _Float16* op = out + (size_t)row * H_;
  const float4* gp = (const float4*)g;
  const float4* bp = (const float4*)bt;
#pragma unroll
  for (int i = 0; i < 4; ++i) {
    const int c4 = lane + 32 * i;
    float4 gv = gp[c4], bv = bp[c4];
    op[c4 * 4 + 0] = (_Float16)((v[i].x - mean) * inv * gv.x + bv.x);
    op[c4 * 4 + 1] = (_Float16)((v[i].y - mean) * inv * gv.y + bv.y);
    op[c4 * 4 + 2] = (_Float16)((v[i].z - mean) * inv * gv.z + bv.z);
    op[c4 * 4 + 3] = (_Float16)((v[i].w - mean) * inv * gv.w + bv.w);
  }
}

// ---------------------------------------------------------------------------
// Weight convert+transpose via LDS tile: w (Kin x Nout) f32 -> wt (Nout x Kin) f16
// grid (Kin/32, Nout/32), block 256 (32x8). Coalesced on both sides.
// ---------------------------------------------------------------------------
__global__ __launch_bounds__(256) void wtrans_kernel(
    const float* __restrict__ w, _Float16* __restrict__ wt, int Kin, int Nout) {
  __shared__ float tile[32][33];
  const int tx = threadIdx.x & 31;
  const int ty = threadIdx.x >> 5;       // 0..7
  const int i0 = blockIdx.x * 32;        // Kin base
  const int o0 = blockIdx.y * 32;        // Nout base
#pragma unroll
  for (int s = 0; s < 4; ++s)
    tile[ty + s * 8][tx] = w[(size_t)(i0 + ty + s * 8) * Nout + (o0 + tx)];
  __syncthreads();
#pragma unroll
  for (int s = 0; s < 4; ++s)
    wt[(size_t)(o0 + ty + s * 8) * Kin + (i0 + tx)] =
        (_Float16)tile[tx][ty + s * 8];
}

// ---------------------------------------------------------------------------
// GEMM: C(MxN) = A(MxK,f16,rm) * Bt(NxK,f16,rm)^T + bias
// Block 256 thr = 8 waves (4 m x 2 n). Block tile 128x128, wave tile 32x64
// (2x4 WMMA frags, 8 wmma per k-step). B tile (128 rows x 64 K) staged to LDS
// by the Tensor Data Mover, double-buffered; A streamed from global (L2-hot).
// All 4 B fragments are loaded into distinct registers BEFORE the 8 WMMAs so
// the ds_load_b128s pipeline under the matrix ops instead of s_wait_dscnt 0
// stalls between every WMMA pair.
// mode 0: f16 store (+bias); 1: f16 gelu(+bias); 2: fp32 residual +=
// ---------------------------------------------------------------------------
__global__ __launch_bounds__(256) void gemm_wmma_kernel(
    const _Float16* __restrict__ A, const _Float16* __restrict__ Bt,
    const float* __restrict__ bias, _Float16* __restrict__ outh,
    float* __restrict__ outf, int M, int N, int K, int mode) {
  __shared__ __align__(16) _Float16 bsm[2][128 * 64];   // 2 x 16KB
  const int lane = threadIdx.x & 31;
  const int wave = threadIdx.x >> 5;
  const int mw = wave & 3;             // 4 m-waves
  const int nw = wave >> 2;            // 2 n-waves
  const int r = lane & 15;

  const int m0 = blockIdx.x * 128 + mw * 32;
  const int nblk = blockIdx.y * 128;   // block column base
  const int n0w = nw * 64;             // within-block column base

  // Static part of the tensor descriptor (group 1).
  u32x8 g1;
  g1[0] = 0x00010000u;                                        // data_size=2B
  g1[1] = ((unsigned)K & 0xFFFFu) << 16;                      // tensor_dim0 lo
  g1[2] = ((unsigned)K >> 16) | (((unsigned)N & 0xFFFFu) << 16); // dim0 hi|dim1 lo
  g1[3] = ((unsigned)N >> 16) | (64u << 16);                  // dim1 hi | tile_dim0=64
  g1[4] = 128u;                                               // tile_dim1=128, tile_dim2=0
  g1[5] = (unsigned)K;                                        // tensor_dim0_stride lo
  g1[6] = 0u;
  g1[7] = 0u;

  v8f acc[2][4] = {};
  const _Float16* a0p = A + (size_t)(m0 + r) * K;
  const _Float16* a1p = A + (size_t)(m0 + 16 + r) * K;
  const int nch = K >> 6;              // K chunks of 64

  if (wave == 0) {
    tdm_load_to_lds(tdm_g0((unsigned)(size_t)&bsm[0][0],
                           Bt + (size_t)nblk * K), g1);
    __builtin_amdgcn_s_wait_tensorcnt(0);
  }
  __syncthreads();

  for (int j = 0; j < nch; ++j) {
    const int k0 = j << 6;
    if (wave == 0 && j + 1 < nch) {
      tdm_load_to_lds(tdm_g0((unsigned)(size_t)&bsm[(j + 1) & 1][0],
                             Bt + (size_t)nblk * K + (k0 + 64)), g1);
    }
    const _Float16* bs = bsm[j & 1];
    __builtin_prefetch(a0p + k0 + 64, 0, 1);   // next A chunk
    __builtin_prefetch(a1p + k0 + 64, 0, 1);
#pragma unroll
    for (int ks = 0; ks < 2; ++ks) {
      const int kk = k0 + ks * 32;
      v16h a0 = frag_ld(a0p + kk);
      v16h a1 = frag_ld(a1p + kk);
      v16h bfr[4];
#pragma unroll
      for (int fc = 0; fc < 4; ++fc)
        bfr[fc] = frag_ld(bs + (size_t)(n0w + fc * 16 + r) * 64 + ks * 32);
#pragma unroll
      for (int fc = 0; fc < 4; ++fc) {
        acc[0][fc] = wmma32(a0, bfr[fc], acc[0][fc]);
        acc[1][fc] = wmma32(a1, bfr[fc], acc[1][fc]);
      }
    }
    if (wave == 0) __builtin_amdgcn_s_wait_tensorcnt(0);
    __syncthreads();
  }

  const int rowoff = (lane & 16) ? 8 : 0;
#pragma unroll
  for (int fr = 0; fr < 2; ++fr)
#pragma unroll
    for (int fc = 0; fc < 4; ++fc) {
      const int nn = nblk + n0w + fc * 16 + r;
      const float bv = bias ? bias[nn] : 0.0f;
#pragma unroll
      for (int i = 0; i < 8; ++i) {
        const int mm = m0 + fr * 16 + rowoff + i;
        float vv = acc[fr][fc][i] + bv;
        if (mode == 1) vv = 0.5f * vv * (1.0f + erff(vv * 0.70710678f));
        const size_t idx = (size_t)mm * N + nn;
        if (mode <= 1) outh[idx] = (_Float16)vv;
        else           outf[idx] += vv;
      }
    }
}

// ---------------------------------------------------------------------------
// Flash attention, causal + key-padding mask.
// Block = 4 waves per (b,head); each wave owns 16-row q-tiles (flash loop).
// ---------------------------------------------------------------------------
__global__ __launch_bounds__(128) void attn_kernel(
    const _Float16* __restrict__ qkv, const int* __restrict__ hist,
    _Float16* __restrict__ ao) {
  __shared__ __align__(16) _Float16 plds[4][16 * 32];
  const int lane = threadIdx.x & 31;
  const int wave = threadIdx.x >> 5;
  const int b = blockIdx.x / NH_;
  const int h = blockIdx.x % NH_;
  const int r = lane & 15;
  const int rowoff = (lane & 16) ? 8 : 0;
  const int kb = (lane & 16) ? 8 : 0;

  const _Float16* qb = qkv + (size_t)b * L_ * H3_ + h * DH_;
  const _Float16* kbp = qb + H_;
  const _Float16* vbp = qb + 2 * H_;
  const int* hb = hist + b * L_;
  _Float16* lp = plds[wave];

  for (int qt = wave; qt < QTILES; qt += 4) {
    const int q0 = qt * 16;
    const int qrowA = min(q0 + r, L_ - 1);
    const v16h qf0 = frag_ld(qb + (size_t)qrowA * H3_ + 0);
    const v16h qf1 = frag_ld(qb + (size_t)qrowA * H3_ + 32);

    float mi[8], li[8];
    v8f acc[4] = {};
#pragma unroll
    for (int i = 0; i < 8; ++i) { mi[i] = -1e30f; li[i] = 0.f; }

    const int kend = q0 + 16;          // causal upper bound on keys
    for (int k0 = 0; k0 < kend; k0 += 32) {
      float p2[2][8];
      // ---- scores: two 16-wide S tiles via WMMA (K = DH = 2x32) ----
#pragma unroll
      for (int st = 0; st < 2; ++st) {
        const int kt = k0 + st * 16;
        const int krow = min(kt + r, L_ - 1);
        v16h kf0 = frag_ld(kbp + (size_t)krow * H3_ + 0);
        v16h kf1 = frag_ld(kbp + (size_t)krow * H3_ + 32);
        v8f s = {};
        s = wmma32(qf0, kf0, s);
        s = wmma32(qf1, kf1, s);
        const int kk = kt + r;         // key column handled by this lane
        const bool kmask = (kk >= L_) || (hb[min(kk, L_ - 1)] == 0);
#pragma unroll
        for (int i = 0; i < 8; ++i) {
          const int qrow = q0 + rowoff + i;
          float sv = s[i] * 0.125f;    // 1/sqrt(64)
          if (kmask || kk > qrow) sv = -1e30f;
          p2[st][i] = sv;
        }
      }
      // ---- online softmax: row max/sum across the 16-lane N-group ----
      float mnew[8];
#pragma unroll
      for (int i = 0; i < 8; ++i) {
        float v = fmaxf(p2[0][i], p2[1][i]);
#pragma unroll
        for (int m = 8; m >= 1; m >>= 1) v = fmaxf(v, __shfl_xor(v, m, 32));
        mnew[i] = fmaxf(mi[i], v);
      }
#pragma unroll
      for (int i = 0; i < 8; ++i) {
        const float alpha = __expf(mi[i] - mnew[i]);
        const float e0 = __expf(p2[0][i] - mnew[i]);
        const float e1 = __expf(p2[1][i] - mnew[i]);
        lp[(rowoff + i) * 32 + r] = (_Float16)e0;
        lp[(rowoff + i) * 32 + 16 + r] = (_Float16)e1;
        float rs = e0 + e1;
#pragma unroll
        for (int m = 8; m >= 1; m >>= 1) rs += __shfl_xor(rs, m, 32);
        li[i] = li[i] * alpha + rs;
        mi[i] = mnew[i];
#pragma unroll
        for (int f = 0; f < 4; ++f) acc[f][i] *= alpha;
      }
      asm volatile("s_wait_dscnt 0" ::: "memory");
      // ---- reload P in A-fragment layout from LDS ----
      const v16h pf = frag_ld(lp + (size_t)r * 32);
      // ---- P(16x32) @ V(32x64) as 4 WMMAs ----
#pragma unroll
      for (int f = 0; f < 4; ++f) {
        v16h vf;
#pragma unroll
        for (int e = 0; e < 8; ++e) {
          const int key0 = min(k0 + kb + e, L_ - 1);
          const int key1 = min(k0 + kb + 16 + e, L_ - 1);
          vf[e]     = vbp[(size_t)key0 * H3_ + f * 16 + r];
          vf[e + 8] = vbp[(size_t)key1 * H3_ + f * 16 + r];
        }
        acc[f] = wmma32(pf, vf, acc[f]);
      }
    }
    // ---- normalize + store f16 ao (row stride H) ----
#pragma unroll
    for (int f = 0; f < 4; ++f)
#pragma unroll
      for (int i = 0; i < 8; ++i) {
        const int q = q0 + rowoff + i;
        if (q < L_)
          ao[((size_t)(b * L_ + q)) * H_ + h * DH_ + f * 16 + r] =
              (_Float16)(acc[f][i] / li[i]);
      }
  }
}

// ---------------------------------------------------------------------------
// Final LN + gather last valid position -> out (B,H) fp32
// ---------------------------------------------------------------------------
__global__ __launch_bounds__(256) void final_kernel(
    const float* __restrict__ x, const int* __restrict__ hist,
    const float* __restrict__ g, const float* __restrict__ bt,
    float* __restrict__ out) {
  __shared__ float red[256];
  __shared__ int slen;
  const int b = blockIdx.x;
  const int t = threadIdx.x;
  int cnt = 0;
  for (int l = t; l < L_; l += 256) cnt += (hist[b * L_ + l] > 0) ? 1 : 0;
  red[t] = (float)cnt;
  __syncthreads();
  for (int s = 128; s >= 1; s >>= 1) {
    if (t < s) red[t] += red[t + s];
    __syncthreads();
  }
  if (t == 0) slen = (int)(red[0] + 0.5f);
  __syncthreads();
  const int last = max(slen - 1, 0);
  const float* xr = x + ((size_t)b * L_ + last) * H_;
  const float v0 = xr[t], v1 = xr[t + 256];
  __syncthreads();
  red[t] = v0 + v1;
  __syncthreads();
  for (int s = 128; s >= 1; s >>= 1) {
    if (t < s) red[t] += red[t + s];
    __syncthreads();
  }
  const float mean = red[0] * (1.0f / H_);
  __syncthreads();
  const float d0 = v0 - mean, d1 = v1 - mean;
  red[t] = d0 * d0 + d1 * d1;
  __syncthreads();
  for (int s = 128; s >= 1; s >>= 1) {
    if (t < s) red[t] += red[t + s];
    __syncthreads();
  }
  const float inv = rsqrtf(red[0] * (1.0f / H_) + 1e-5f);
  out[(size_t)b * H_ + t]       = d0 * inv * g[t] + bt[t];
  out[(size_t)b * H_ + t + 256] = d1 * inv * g[t + 256] + bt[t + 256];
}

// ---------------------------------------------------------------------------
// Host orchestration
// ---------------------------------------------------------------------------
extern "C" void kernel_launch(void* const* d_in, const int* in_sizes, int n_in,
                              void* d_out, int out_size, void* d_ws, size_t ws_size,
                              hipStream_t stream) {
  const int*   hist     = (const int*)d_in[0];
  const float* item_emb = (const float*)d_in[1];
  const float* pos_emb  = (const float*)d_in[2];
  const float* in_w     = (const float*)d_in[3];
  const float* in_b     = (const float*)d_in[4];
  const float* out_w    = (const float*)d_in[5];
  const float* out_b    = (const float*)d_in[6];
  const float* ln1g     = (const float*)d_in[7];
  const float* ln1b     = (const float*)d_in[8];
  const float* ln2g     = (const float*)d_in[9];
  const float* ln2b     = (const float*)d_in[10];
  const float* w1       = (const float*)d_in[11];
  const float* b1       = (const float*)d_in[12];
  const float* w2       = (const float*)d_in[13];
  const float* b2       = (const float*)d_in[14];
  const float* fing     = (const float*)d_in[15];
  const float* finb     = (const float*)d_in[16];
  float* out = (float*)d_out;

  // workspace layout
  char* ws = (char*)d_ws;
  const size_t XB  = (size_t)BL_ * H_ * sizeof(float);        // x, fp32
  const size_t AFB = (size_t)BL_ * H4_ * sizeof(_Float16);    // f16 scratch (4H wide)
  const size_t QKB = (size_t)BL_ * H3_ * sizeof(_Float16);    // qkv f16
  float*    X  = (float*)ws;
  _Float16* AF = (_Float16*)(ws + XB);
  _Float16* QK = (_Float16*)(ws + XB + AFB);
  _Float16* WT = (_Float16*)(ws + XB + AFB + QKB);            // <= 4MB weight tile
  (void)ws_size; (void)n_in; (void)in_sizes; (void)out_size;

  const dim3 blk256(256), blk128(128);

  // 1) embeddings
  embed_kernel<<<BL_, blk128, 0, stream>>>(hist, item_emb, pos_emb, X);

  for (int i = 0; i < NL_; ++i) {
    // --- attention block ---
    ln_kernel<<<BL_ / 8, blk256, 0, stream>>>(X, ln1g + i * H_, ln1b + i * H_, AF, BL_);

    wtrans_kernel<<<dim3(H_ / 32, H3_ / 32), blk256, 0, stream>>>(
        in_w + (size_t)i * H_ * H3_, WT, H_, H3_);
    gemm_wmma_kernel<<<dim3(BL_ / 128, H3_ / 128), blk256, 0, stream>>>(
        AF, WT, in_b + (size_t)i * H3_, QK, nullptr, BL_, H3_, H_, 0);

    attn_kernel<<<B_ * NH_, blk128, 0, stream>>>(QK, hist, AF);

    wtrans_kernel<<<dim3(H_ / 32, H_ / 32), blk256, 0, stream>>>(
        out_w + (size_t)i * H_ * H_, WT, H_, H_);
    gemm_wmma_kernel<<<dim3(BL_ / 128, H_ / 128), blk256, 0, stream>>>(
        AF, WT, out_b + (size_t)i * H_, nullptr, X, BL_, H_, H_, 2);

    // --- FFN block ---
    ln_kernel<<<BL_ / 8, blk256, 0, stream>>>(X, ln2g + i * H_, ln2b + i * H_, QK, BL_);

    wtrans_kernel<<<dim3(H_ / 32, H4_ / 32), blk256, 0, stream>>>(
        w1 + (size_t)i * H_ * H4_, WT, H_, H4_);
    gemm_wmma_kernel<<<dim3(BL_ / 128, H4_ / 128), blk256, 0, stream>>>(
        QK, WT, b1 + (size_t)i * H4_, AF, nullptr, BL_, H4_, H_, 1);

    wtrans_kernel<<<dim3(H4_ / 32, H_ / 32), blk256, 0, stream>>>(
        w2 + (size_t)i * H4_ * H_, WT, H4_, H_);
    gemm_wmma_kernel<<<dim3(BL_ / 128, H_ / 128), blk256, 0, stream>>>(
        AF, WT, b2 + (size_t)i * H_, nullptr, X, BL_, H_, H4_, 2);
  }

  // 3) final LN + gather
  final_kernel<<<B_, blk256, 0, stream>>>(X, hist, fing, finb, out);
}
